// att_Module_58918361366632
// MI455X (gfx1250) — matrix operand order repaired and checked by reading, not compiled, not run on hardware
//
#include <hip/hip_runtime.h>
#include <hip/hip_bf16.h>
#include <math.h>

// Problem constants
#define BB   64
#define NN   4096
#define DD   512
#define AT   128
#define TILE_N   128
#define CHUNKS   (NN / TILE_N)      // 32
#define KSTEPS   (DD / 32)          // 16
#define ROWPITCH 1040               // 512*2 bytes bf16 + 16B pad (bank-conflict break)

typedef __attribute__((ext_vector_type(16))) __bf16 v16bf;
typedef __attribute__((ext_vector_type(8)))  float  v8f;
typedef __attribute__((ext_vector_type(2)))  __bf16 bf16x2;

// Hardware tanh (TRANS op on gfx1250) when the builtin exists.
#if defined(__has_builtin)
#if __has_builtin(__builtin_amdgcn_tanhf)
#define FAST_TANH(x) __builtin_amdgcn_tanhf(x)
#endif
#endif
#ifndef FAST_TANH
#define FAST_TANH(x) tanhf(x)
#endif

// Scheduling pin: DS-read group / WMMA group
#if defined(__has_builtin)
#if __has_builtin(__builtin_amdgcn_sched_group_barrier)
#define SGB(mask, cnt) __builtin_amdgcn_sched_group_barrier((mask), (cnt), 0)
#endif
#endif
#ifndef SGB
#define SGB(mask, cnt)
#endif

__device__ __forceinline__ unsigned pack_bf16(float x, float y) {
    union { bf16x2 v; unsigned u; } q;
    q.v[0] = (__bf16)x;
    q.v[1] = (__bf16)y;
    return q.u;
}

__device__ __forceinline__ float lo_bf16(unsigned u) {
    union { unsigned u; bf16x2 v; } q; q.u = u; return (float)q.v[0];
}
__device__ __forceinline__ float hi_bf16(unsigned u) {
    union { unsigned u; bf16x2 v; } q; q.u = u; return (float)q.v[1];
}

// ---------------------------------------------------------------------------
// K0a: vq_att[b][a] = tanh(v_Q[b,:] . W_Q[:,a] + b_Q[a])      grid=B, block=AT
// ---------------------------------------------------------------------------
__global__ void k_prep_q(const float* __restrict__ vQ, const float* __restrict__ WQ,
                         const float* __restrict__ bQ, float* __restrict__ vq_att) {
    int b = blockIdx.x, a = threadIdx.x;
    float s = bQ[a];
    for (int d = 0; d < DD; ++d)
        s += vQ[b * DD + d] * WQ[d * AT + a];
    vq_att[b * AT + a] = FAST_TANH(s);
}

// ---------------------------------------------------------------------------
// K0b: swizzle W_I (f32 [D][AT]) into WMMA-B bf16 per-lane layout.
// total 32768 dwords -> grid=128, block=256
// ---------------------------------------------------------------------------
__global__ void k_prep_wi(const float* __restrict__ WI, unsigned* __restrict__ out) {
    int t = blockIdx.x * 256 + threadIdx.x;
    int j  = t & 7;
    int l  = (t >> 3) & 31;
    int c  = (t >> 8) & 7;
    int ks = t >> 11;
    int k0  = ks * 32 + ((l & 16) ? 8 : 0) + ((j & 4) ? 16 : 0) + 2 * (j & 3);
    int col = c * 16 + (l & 15);
    float lo = WI[k0 * AT + col];
    float hi = WI[(k0 + 1) * AT + col];
    out[t] = pack_bf16(lo, hi);
}

// ---------------------------------------------------------------------------
// K1: fused  logits -> chunk softmax -> weighted v_I accumulation
// grid = B*CHUNKS (2048), block = 256 (8 waves), dyn LDS = 128*ROWPITCH
// ---------------------------------------------------------------------------
__global__ __launch_bounds__(256, 2)
void k_fused(const float* __restrict__ vI, const float* __restrict__ vq_att,
             const uint4* __restrict__ wiB, const float* __restrict__ Wp,
             const float* __restrict__ bp,
             float* __restrict__ m_arr, float* __restrict__ s_arr,
             float* __restrict__ acc_out, float* __restrict__ wexp_out) {
    extern __shared__ __align__(16) char smemA[];       // 128 x 512 bf16, pitch 1040
    __shared__ float partial[8][TILE_N];
    __shared__ float wtile[TILE_N];
    __shared__ float red[TILE_N];

    const int b     = blockIdx.x >> 5;
    const int chunk = blockIdx.x & 31;
    const int n0    = chunk * TILE_N;
    const int t     = threadIdx.x;
    const int wave  = t >> 5;
    const int lane  = t & 31;

    // ---- Stage v_I tile (contiguous 256 KiB) -> LDS as bf16, coalesced ----
    const float4* src = (const float4*)(vI + (size_t)(b * NN + n0) * DD);
    #pragma unroll 4
    for (int it = 0; it < 64; ++it) {
        int e4 = t + it * 256;                 // float4 index within tile
        float4 f = src[e4];
        int e   = e4 * 4;
        int row = e >> 9;                      // /512
        int k   = e & 511;
        uint2 pk;
        pk.x = pack_bf16(f.x, f.y);
        pk.y = pack_bf16(f.z, f.w);
        *(uint2*)(smemA + row * ROWPITCH + k * 2) = pk;
    }

    // ---- Load resident B panel: this wave's 16 columns, all 16 K-steps ----
    v16bf Breg[KSTEPS];                        // 128 VGPRs
    #pragma unroll
    for (int ks = 0; ks < KSTEPS; ++ks) {
        union { uint4 q[2]; v16bf v; } tmp;
        int idx = (((ks * 8 + wave) * 32) + lane) * 2;
        tmp.q[0] = wiB[idx];
        tmp.q[1] = wiB[idx + 1];
        Breg[ks] = tmp.v;
    }

    const int   colLane = lane & 15;
    const float vqa = vq_att[b * AT + wave * 16 + colLane];
    const float wpv = Wp[wave * 16 + colLane];

    __syncthreads();

    // ---- 8 row-chunks x 16 K-steps of WMMA (A loads pipelined 4 deep) ----
    union AF { uint4 q[2]; v16bf v; };
    #pragma unroll
    for (int m = 0; m < 8; ++m) {
        const char* abase = smemA + (m * 16 + colLane) * ROWPITCH
                          + ((lane & 16) ? 16 : 0);
        AF a[4];
        #pragma unroll
        for (int p = 0; p < 4; ++p) {
            a[p].q[0] = *(const uint4*)(abase + p * 64);
            a[p].q[1] = *(const uint4*)(abase + p * 64 + 32);
        }
        v8f c = {};
        #pragma unroll
        for (int ks = 0; ks < KSTEPS; ++ks) {
            c = __builtin_amdgcn_wmma_f32_16x16x32_bf16(
                    false, a[ks & 3].v, false, Breg[ks], (short)0, c, false, false);
            if (ks + 4 < KSTEPS) {
                a[ks & 3].q[0] = *(const uint4*)(abase + (ks + 4) * 64);
                a[ks & 3].q[1] = *(const uint4*)(abase + (ks + 4) * 64 + 32);
            }
        }
        // Pin schedule: 8 DS reads, then (1 WMMA, 2 DS reads) x12, then 4 WMMAs.
        SGB(0x100, 8);
        #pragma unroll
        for (int g = 0; g < 12; ++g) {
            SGB(0x008, 1);
            SGB(0x100, 2);
        }
        SGB(0x008, 4);

        // h = tanh(tanh(c)+vq); contribution Wp*h; reduce across 16-lane halves
        #pragma unroll
        for (int r = 0; r < 8; ++r) {
            float h = wpv * FAST_TANH(FAST_TANH(c[r]) + vqa);
            h += __shfl_xor(h, 1);
            h += __shfl_xor(h, 2);
            h += __shfl_xor(h, 4);
            h += __shfl_xor(h, 8);
            if (lane == r)      partial[wave][m * 16 + r] = h;
            if (lane == 16 + r) partial[wave][m * 16 + 8 + r] = h;
        }
    }
    __syncthreads();

    // ---- logits, chunk-local online softmax ----
    float logit = 0.f;
    if (t < TILE_N) {
        logit = bp[0];
        #pragma unroll
        for (int w = 0; w < 8; ++w) logit += partial[w][t];
        red[t] = logit;
    }
    __syncthreads();
    for (int s = 64; s > 0; s >>= 1) {
        if (t < s) red[t] = fmaxf(red[t], red[t + s]);
        __syncthreads();
    }
    const float mC = red[0];
    __syncthreads();
    if (t < TILE_N) {
        float we = expf(logit - mC);
        wtile[t] = we;
        wexp_out[(size_t)b * NN + n0 + t] = we;
        red[t] = we;
    }
    __syncthreads();
    for (int s = 64; s > 0; s >>= 1) {
        if (t < s) red[t] += red[t + s];
        __syncthreads();
    }
    const float sC = red[0];
    if (t == 0) { m_arr[b * CHUNKS + chunk] = mC; s_arr[b * CHUNKS + chunk] = sC; }

    // ---- weighted accumulation of LDS bf16 tile: acc[d] = sum_r w_r * v_I ----
    float a0 = 0.f, a1 = 0.f;
    const int d2 = t * 2;
    #pragma unroll 2
    for (int r4 = 0; r4 < TILE_N / 4; ++r4) {
        float4 w4 = *(const float4*)&wtile[r4 * 4];        // 1 ds_load_b128 / 4 rows
        #pragma unroll
        for (int i = 0; i < 4; ++i) {
            int r = r4 * 4 + i;
            float w = (i == 0) ? w4.x : (i == 1) ? w4.y : (i == 2) ? w4.z : w4.w;
            unsigned pk = *(const unsigned*)(smemA + r * ROWPITCH + d2 * 2);
            a0 += w * lo_bf16(pk);
            a1 += w * hi_bf16(pk);
        }
    }
    float* accp = acc_out + ((size_t)(b * CHUNKS + chunk)) * DD;
    accp[d2]     = a0;
    accp[d2 + 1] = a1;
}

// ---------------------------------------------------------------------------
// K2: combine 32 chunk partials per batch -> u = acc/s + v_Q.  grid=B, block=256
// ---------------------------------------------------------------------------
__global__ __launch_bounds__(256)
void k_reduce(const float* __restrict__ m_arr, const float* __restrict__ s_arr,
              const float* __restrict__ acc, const float* __restrict__ vQ,
              float* __restrict__ msg, float* __restrict__ out_u) {
    __shared__ float fac[CHUNKS];
    __shared__ float sg_sh;
    int b = blockIdx.x, t = threadIdx.x;
    if (t < CHUNKS) {
        float m  = m_arr[b * CHUNKS + t];
        float mm = m;
        for (int o = 1; o < 32; o <<= 1) mm = fmaxf(mm, __shfl_xor(mm, o));
        float f  = expf(m - mm);
        float sp = s_arr[b * CHUNKS + t] * f;
        float ss = sp;
        for (int o = 1; o < 32; o <<= 1) ss += __shfl_xor(ss, o);
        fac[t] = f;
        if (t == 0) {
            sg_sh = ss;
            msg[b * 2]     = mm;
            msg[b * 2 + 1] = ss;
        }
    }
    __syncthreads();
    float inv = 1.0f / sg_sh;
    #pragma unroll
    for (int half = 0; half < 2; ++half) {
        int d = t + half * 256;
        float s = 0.f;
        for (int j = 0; j < CHUNKS; ++j)
            s += fac[j] * acc[((size_t)(b * CHUNKS + j)) * DD + d];
        out_u[b * DD + d] = s * inv + vQ[b * DD + d];
    }
}

// ---------------------------------------------------------------------------
// K3: p[b,n] = wexp[b,n] * exp(m_chunk - m_g) / s_g.   grid=B*N/256, block=256
// ---------------------------------------------------------------------------
__global__ void k_norm(const float* __restrict__ wexp, const float* __restrict__ m_arr,
                       const float* __restrict__ msg, float* __restrict__ out_p) {
    int i = blockIdx.x * 256 + threadIdx.x;
    int b     = i >> 12;                  // /4096
    int chunk = (i & (NN - 1)) >> 7;      // /128
    float mg = msg[b * 2];
    float sg = msg[b * 2 + 1];
    out_p[i] = wexp[i] * expf(m_arr[b * CHUNKS + chunk] - mg) / sg;
}

// ---------------------------------------------------------------------------
extern "C" void kernel_launch(void* const* d_in, const int* in_sizes, int n_in,
                              void* d_out, int out_size, void* d_ws, size_t ws_size,
                              hipStream_t stream) {
    const float* vI = (const float*)d_in[0];
    const float* vQ = (const float*)d_in[1];
    const float* WI = (const float*)d_in[2];
    const float* WQ = (const float*)d_in[3];
    const float* bQ = (const float*)d_in[4];
    const float* Wp = (const float*)d_in[5];
    const float* bp = (const float*)d_in[6];

    float* out_p = (float*)d_out;              // [B,N]
    float* out_u = out_p + (size_t)BB * NN;    // [B,D]

    // workspace layout (floats)
    float* ws       = (float*)d_ws;
    float* vq_att   = ws;                                   //  B*AT          = 8192
    unsigned* wiB   = (unsigned*)(ws + 8192);               //  32768 dwords (bf16 W_I)
    float* m_arr    = ws + 8192 + 32768;                    //  B*CHUNKS      = 2048
    float* s_arr    = m_arr + BB * CHUNKS;                  //  2048
    float* acc_ws   = s_arr + BB * CHUNKS;                  //  B*CHUNKS*D    = 1048576
    float* wexp_ws  = acc_ws + (size_t)BB * CHUNKS * DD;    //  B*N           = 262144
    float* msg      = wexp_ws + (size_t)BB * NN;            //  B*2

    k_prep_q<<<BB, AT, 0, stream>>>(vQ, WQ, bQ, vq_att);
    k_prep_wi<<<128, 256, 0, stream>>>(WI, wiB);

    const size_t smem = (size_t)TILE_N * ROWPITCH;          // 133,120 B
    k_fused<<<BB * CHUNKS, 256, smem, stream>>>(vI, vq_att, (const uint4*)wiB,
                                                Wp, bp, m_arr, s_arr, acc_ws, wexp_ws);

    k_reduce<<<BB, 256, 0, stream>>>(m_arr, s_arr, acc_ws, vQ, msg, out_u);
    k_norm<<<(BB * NN) / 256, 256, 0, stream>>>(wexp_ws, m_arr, msg, out_p);
}
